// MultiHeadAttention_27178553049340
// MI455X (gfx1250) — compile-verified, matrix operand order checked by reading, not run on hardware
//
#include <hip/hip_runtime.h>
#include <hip/hip_bf16.h>

// MI455X (gfx1250) wave32 WMMA implementation of T5-style MHA.
// All GEMMs (QKV proj, QK^T, PV, out proj) run on v_wmma_f32_16x16x32_bf16.
// bf16 global->LDS staging uses CDNA5 global_load_async_to_lds_b128 (ASYNCcnt)
// with explicit double-buffering: tile i+1's copies are issued under tile i's WMMAs.

#define D_MODEL 1472
#define N_HEADS 6
#define D_KV    64
#define INNER   384
#define SEQ     2048
#define BATCH   4
#define BIAS_LEN (2 * SEQ - 1)   // 4095

typedef __attribute__((ext_vector_type(16))) __bf16 v16bf;
typedef __attribute__((ext_vector_type(8)))  __bf16 v8bf;
typedef __attribute__((ext_vector_type(4)))  __bf16 v4bf;
typedef __attribute__((ext_vector_type(8)))  float  v8f;

static __device__ __forceinline__ v8f wmma_bf16(v16bf a, v16bf b, v8f c) {
  // (neg_a, A, neg_b, B, c_mod, C, reuse_a, reuse_b)
  return __builtin_amdgcn_wmma_f32_16x16x32_bf16(false, a, false, b, (short)0, c, false, false);
}

// CDNA5 async global->LDS copy, 16B per lane, tracked by ASYNCcnt.
// No memory clobber on issue: volatile asm keeps issue->wait order; normal
// loads/stores may still be scheduled around it.
static __device__ __forceinline__ void async_load_b128(const void* gptr, void* lptr) {
  unsigned lds = (unsigned)(uintptr_t)lptr;               // flat LDS addr: low 32b = LDS offset
  unsigned long long ga = (unsigned long long)(uintptr_t)gptr;
  asm volatile("global_load_async_to_lds_b128 %0, %1, off" :: "v"(lds), "v"(ga));
}
static __device__ __forceinline__ void wait_async() {
  asm volatile("s_wait_asynccnt 0x0" ::: "memory");
}

// A/B fragment loader per ISA 7.12.2 (16-bit A 16x32):
// lanes 0-15 hold rows 0-15 with K=0..7 (vgpr0-3) and K=16..23 (vgpr4-7);
// lanes 16-31 hold rows 0-15 with K=8..15 and K=24..31.
static __device__ __forceinline__ v16bf load_frag(const __bf16* base, int row, int ld,
                                                  int k0, int lane) {
  const int hi = (lane >> 4) & 1;
  const int r  = row + (lane & 15);
  const __bf16* p = base + (size_t)r * ld + k0 + hi * 8;
  union { v16bf v; v8bf h[2]; } u;
  u.h[0] = *(const v8bf*)(p);
  u.h[1] = *(const v8bf*)(p + 16);
  return u.v;
}

static __device__ __forceinline__ v8f vzero8() {
  v8f z = {0.f, 0.f, 0.f, 0.f, 0.f, 0.f, 0.f, 0.f};
  return z;
}

// ---------------------------------------------------------------------------
// prep: W [K][N] f32 -> Wt [N][K] bf16
// ---------------------------------------------------------------------------
__global__ void k_transpose_bf16(const float* __restrict__ W, __bf16* __restrict__ Wt,
                                 int K, int N) {
  int total = K * N;
  for (int idx = blockIdx.x * blockDim.x + threadIdx.x; idx < total;
       idx += gridDim.x * blockDim.x) {
    int n = idx % N;
    int k = idx / N;
    Wt[(size_t)n * K + k] = (__bf16)W[(size_t)k * N + n];
  }
}

// T5 relative-position bias as 1D table: table[h][i], i = (k - q) + (SEQ-1)
__global__ void k_build_bias(const float* __restrict__ rel_bias, float* __restrict__ table) {
  int idx = blockIdx.x * blockDim.x + threadIdx.x;
  if (idx >= N_HEADS * BIAS_LEN) return;
  int h = idx / BIAS_LEN;
  int i = idx % BIAS_LEN;
  int d = i - (SEQ - 1);            // relative position (mem - ctx)
  const int nb = 16;                // 32 buckets, bidirectional -> 16
  int ret = (d > 0) ? nb : 0;
  int rel = (d < 0) ? -d : d;
  const int max_exact = 8;
  int bucket;
  if (rel < max_exact) {
    bucket = ret + rel;
  } else {
    float relf = (float)rel;
    int large = max_exact + (int)(__logf(relf / 8.0f) / __logf(16.0f) * 8.0f);
    if (large > nb - 1) large = nb - 1;
    bucket = ret + large;
  }
  table[h * BIAS_LEN + i] = rel_bias[bucket * N_HEADS + h];
}

// ---------------------------------------------------------------------------
// QKV projection: X[8192x1472] f32 @ Wt[384x1472] bf16 -> {Q,K,V}[B,H,S,64] bf16
// 256 threads = 8 waves; 128x64 tile/WG; wave = 32x32 (2x2 accums).
// X tile staged via VGPRs (f32->bf16 cvt); W tile via async-to-LDS; both double-buffered.
// ---------------------------------------------------------------------------
__global__ __launch_bounds__(256) void k_proj_qkv(
    const float* __restrict__ Xq, const float* __restrict__ Xkv,
    const __bf16* __restrict__ WQt, const __bf16* __restrict__ WKt,
    const __bf16* __restrict__ WVt,
    __bf16* __restrict__ Qb, __bf16* __restrict__ Kb, __bf16* __restrict__ Vb) {
  __shared__ __bf16 Xs[2][128][40];   // 32 K + 8 pad halves
  __shared__ __bf16 Ws[2][64][40];

  const int z = blockIdx.z;
  const float*  X   = (z == 0) ? Xq : Xkv;
  const __bf16* Wt  = (z == 0) ? WQt : (z == 1 ? WKt : WVt);
  __bf16*       Out = (z == 0) ? Qb  : (z == 1 ? Kb  : Vb);

  const int m0 = blockIdx.x * 128;
  const int n0 = blockIdx.y * 64;
  const int tid = threadIdx.x;
  const int lane = tid & 31;
  const int w = tid >> 5;
  const int wm = (w & 3) * 32;
  const int wn = (w >> 2) * 32;

  v8f acc[2][2] = {{vzero8(), vzero8()}, {vzero8(), vzero8()}};

  auto stage = [&](int k0, int buf) {
    {  // async-stage W tile 64x32 bf16: one 16B async copy per thread
      int r = tid >> 2;
      int kk = (tid & 3) * 8;
      async_load_b128(Wt + (size_t)(n0 + r) * D_MODEL + k0 + kk, &Ws[buf][r][kk]);
    }
    #pragma unroll
    for (int i = 0; i < 4; ++i) {  // X tile 128x32 f32->bf16
      int g = tid + i * 256;
      int r = g >> 3;
      int kk = (g & 7) * 4;
      float4 xv = *(const float4*)(X + (size_t)(m0 + r) * D_MODEL + k0 + kk);
      v4bf pk = {(__bf16)xv.x, (__bf16)xv.y, (__bf16)xv.z, (__bf16)xv.w};
      *(v4bf*)&Xs[buf][r][kk] = pk;
    }
  };

  const int NT = D_MODEL / 32;   // 46
  stage(0, 0);
  for (int it = 0; it < NT; ++it) {
    const int cur = it & 1;
    wait_async();
    __syncthreads();                       // buf `cur` fully staged by all threads
    if (it + 1 < NT) stage((it + 1) * 32, cur ^ 1);   // overlap with WMMAs below

    v16bf a0 = load_frag(&Xs[cur][0][0], wm + 0,  40, 0, lane);
    v16bf a1 = load_frag(&Xs[cur][0][0], wm + 16, 40, 0, lane);
    v16bf b0 = load_frag(&Ws[cur][0][0], wn + 0,  40, 0, lane);
    v16bf b1 = load_frag(&Ws[cur][0][0], wn + 16, 40, 0, lane);
    acc[0][0] = wmma_bf16(a0, b0, acc[0][0]);
    acc[0][1] = wmma_bf16(a0, b1, acc[0][1]);
    acc[1][0] = wmma_bf16(a1, b0, acc[1][0]);
    acc[1][1] = wmma_bf16(a1, b1, acc[1][1]);
  }

  // scatter to [B,H,S,64] bf16
  const int hi = lane >> 4, ln = lane & 15;
  #pragma unroll
  for (int mi = 0; mi < 2; ++mi) {
    #pragma unroll
    for (int nj = 0; nj < 2; ++nj) {
      int n = n0 + wn + nj * 16 + ln;
      int h = n >> 6, d = n & 63;
      #pragma unroll
      for (int j = 0; j < 8; ++j) {
        int m = m0 + wm + mi * 16 + j + hi * 8;
        int b = m >> 11, s = m & (SEQ - 1);
        Out[(((size_t)b * N_HEADS + h) * SEQ + s) * D_KV + d] = (__bf16)acc[mi][nj][j];
      }
    }
  }
}

// ---------------------------------------------------------------------------
// Flash attention: one WG per (b, h, 64-row q block); 4 waves, wave = 16 q rows.
// K block async-to-LDS, V block transposed through VGPRs; both double-buffered
// so block kb+1 streams in under block kb's WMMAs/softmax.
// ---------------------------------------------------------------------------
__global__ __launch_bounds__(128) void k_attn(
    const __bf16* __restrict__ Qb, const __bf16* __restrict__ Kb,
    const __bf16* __restrict__ Vb, const float* __restrict__ mask,
    const float* __restrict__ bias1d, __bf16* __restrict__ AO) {
  __shared__ __bf16 Ks[2][64][72];   // K block  [kpos][d]
  __shared__ __bf16 Vts[2][64][72];  // V block transposed [d][kpos]
  __shared__ float  Ssm[64][68];     // f32 scores
  __shared__ __bf16 Ps[64][72];      // exp probs, bf16
  __shared__ float  m_s[64], l_s[64], alpha_s[64];

  const int qb = blockIdx.x;        // 0..31
  const int h  = blockIdx.y;        // 0..5
  const int b  = blockIdx.z;        // 0..3
  const int tid = threadIdx.x;
  const int lane = tid & 31;
  const int w = tid >> 5;           // 0..3
  const int hi = lane >> 4, ln = lane & 15;

  const __bf16* Qh = Qb + ((size_t)b * N_HEADS + h) * SEQ * D_KV;
  const __bf16* Kh = Kb + ((size_t)b * N_HEADS + h) * SEQ * D_KV;
  const __bf16* Vh = Vb + ((size_t)b * N_HEADS + h) * SEQ * D_KV;
  const float* biash = bias1d + h * BIAS_LEN;
  const float* maskb = mask + (size_t)b * SEQ * SEQ;

  if (tid < 64) { m_s[tid] = -3.0e38f; l_s[tid] = 0.f; alpha_s[tid] = 1.f; }

  // Q fragments for this wave (persistent)
  const v16bf aq0 = load_frag(Qh, qb * 64 + w * 16, D_KV, 0,  lane);
  const v16bf aq1 = load_frag(Qh, qb * 64 + w * 16, D_KV, 32, lane);

  v8f o[4] = {vzero8(), vzero8(), vzero8(), vzero8()};

  auto stage_kv = [&](int kb, int buf) {
    int kr = tid >> 1;
    int dh = (tid & 1) * 32;
    const __bf16* krow = Kh + ((size_t)(kb * 64 + kr)) * D_KV + dh;
    async_load_b128(krow + 0,  &Ks[buf][kr][dh + 0]);
    async_load_b128(krow + 8,  &Ks[buf][kr][dh + 8]);
    async_load_b128(krow + 16, &Ks[buf][kr][dh + 16]);
    async_load_b128(krow + 24, &Ks[buf][kr][dh + 24]);
    const __bf16* vrow = Vh + ((size_t)(kb * 64 + kr)) * D_KV + dh;
    v8bf v0 = *(const v8bf*)(vrow + 0);
    v8bf v1 = *(const v8bf*)(vrow + 8);
    v8bf v2 = *(const v8bf*)(vrow + 16);
    v8bf v3 = *(const v8bf*)(vrow + 24);
    #pragma unroll
    for (int i = 0; i < 8; ++i) {
      Vts[buf][dh + 0  + i][kr] = v0[i];
      Vts[buf][dh + 8  + i][kr] = v1[i];
      Vts[buf][dh + 16 + i][kr] = v2[i];
      Vts[buf][dh + 24 + i][kr] = v3[i];
    }
  };

  const int NB = SEQ / 64;   // 32
  stage_kv(0, 0);
  for (int kb = 0; kb < NB; ++kb) {
    const int cur = kb & 1;
    wait_async();
    __syncthreads();                      // K/V block `cur` staged; stats visible
    if (kb + 1 < NB) stage_kv(kb + 1, cur ^ 1);   // overlap next block

    // logits: wave rows [w*16, +16) x 64 kpos, fused bias + mask at store
    #pragma unroll
    for (int nj = 0; nj < 4; ++nj) {
      v16bf bk0 = load_frag(&Ks[cur][0][0], nj * 16, 72, 0,  lane);
      v16bf bk1 = load_frag(&Ks[cur][0][0], nj * 16, 72, 32, lane);
      v8f s = vzero8();
      s = wmma_bf16(aq0, bk0, s);
      s = wmma_bf16(aq1, bk1, s);
      int kg = kb * 64 + nj * 16 + ln;
      #pragma unroll
      for (int j = 0; j < 8; ++j) {
        int qr = w * 16 + j + hi * 8;
        int qg = qb * 64 + qr;
        float val = s[j] + biash[kg - qg + (SEQ - 1)] + maskb[(size_t)qg * SEQ + kg];
        Ssm[qr][nj * 16 + ln] = val;
      }
    }
    __syncthreads();

    // online softmax, one thread per q row
    if (tid < 64) {
      float mold = m_s[tid];
      float mx = mold;
      #pragma unroll 4
      for (int j = 0; j < 64; ++j) mx = fmaxf(mx, Ssm[tid][j]);
      float alpha = __expf(mold - mx);
      float sum = 0.f;
      #pragma unroll 4
      for (int j = 0; j < 64; ++j) {
        float p = __expf(Ssm[tid][j] - mx);
        Ps[tid][j] = (__bf16)p;
        sum += p;
      }
      l_s[tid] = l_s[tid] * alpha + sum;
      m_s[tid] = mx;
      alpha_s[tid] = alpha;
    }
    __syncthreads();

    // rescale O accumulators and accumulate P @ V
    float al[8];
    #pragma unroll
    for (int j = 0; j < 8; ++j) al[j] = alpha_s[w * 16 + j + hi * 8];
    #pragma unroll
    for (int nj = 0; nj < 4; ++nj)
      #pragma unroll
      for (int j = 0; j < 8; ++j) o[nj][j] *= al[j];

    v16bf ap0 = load_frag(&Ps[0][0], w * 16, 72, 0,  lane);
    v16bf ap1 = load_frag(&Ps[0][0], w * 16, 72, 32, lane);
    #pragma unroll
    for (int nj = 0; nj < 4; ++nj) {
      v16bf bv0 = load_frag(&Vts[cur][0][0], nj * 16, 72, 0,  lane);
      v16bf bv1 = load_frag(&Vts[cur][0][0], nj * 16, 72, 32, lane);
      o[nj] = wmma_bf16(ap0, bv0, o[nj]);
      o[nj] = wmma_bf16(ap1, bv1, o[nj]);
    }
  }
  __syncthreads();

  // finalize: divide by l, store bf16 attn-out [B,S,INNER]
  float li[8];
  #pragma unroll
  for (int j = 0; j < 8; ++j) li[j] = 1.0f / l_s[w * 16 + j + hi * 8];
  #pragma unroll
  for (int nj = 0; nj < 4; ++nj) {
    int d = nj * 16 + ln;
    #pragma unroll
    for (int j = 0; j < 8; ++j) {
      int qg = qb * 64 + w * 16 + j + hi * 8;
      AO[((size_t)b * SEQ + qg) * INNER + h * D_KV + d] = (__bf16)(o[nj][j] * li[j]);
    }
  }
}

// ---------------------------------------------------------------------------
// Output projection: AO[8192x384] bf16 @ WOt[1472x384] bf16 -> out f32 [8192x1472]
// Both tiles async-to-LDS, double-buffered.
// ---------------------------------------------------------------------------
__global__ __launch_bounds__(256) void k_proj_out(
    const __bf16* __restrict__ A, const __bf16* __restrict__ WOt,
    float* __restrict__ out) {
  __shared__ __bf16 As[2][128][40];
  __shared__ __bf16 Ws[2][64][40];

  const int m0 = blockIdx.x * 128;
  const int n0 = blockIdx.y * 64;
  const int tid = threadIdx.x;
  const int lane = tid & 31;
  const int w = tid >> 5;
  const int wm = (w & 3) * 32;
  const int wn = (w >> 2) * 32;

  v8f acc[2][2] = {{vzero8(), vzero8()}, {vzero8(), vzero8()}};

  auto stage = [&](int k0, int buf) {
    #pragma unroll
    for (int i = 0; i < 2; ++i) {
      int g = tid + i * 256;
      int r = g >> 2;
      int kk = (g & 3) * 8;
      async_load_b128(A + (size_t)(m0 + r) * INNER + k0 + kk, &As[buf][r][kk]);
    }
    {
      int r = tid >> 2;
      int kk = (tid & 3) * 8;
      async_load_b128(WOt + (size_t)(n0 + r) * INNER + k0 + kk, &Ws[buf][r][kk]);
    }
  };

  const int NT = INNER / 32;   // 12
  stage(0, 0);
  for (int it = 0; it < NT; ++it) {
    const int cur = it & 1;
    wait_async();
    __syncthreads();
    if (it + 1 < NT) stage((it + 1) * 32, cur ^ 1);

    v16bf a0 = load_frag(&As[cur][0][0], wm + 0,  40, 0, lane);
    v16bf a1 = load_frag(&As[cur][0][0], wm + 16, 40, 0, lane);
    v16bf b0 = load_frag(&Ws[cur][0][0], wn + 0,  40, 0, lane);
    v16bf b1 = load_frag(&Ws[cur][0][0], wn + 16, 40, 0, lane);
    acc[0][0] = wmma_bf16(a0, b0, acc[0][0]);
    acc[0][1] = wmma_bf16(a0, b1, acc[0][1]);
    acc[1][0] = wmma_bf16(a1, b0, acc[1][0]);
    acc[1][1] = wmma_bf16(a1, b1, acc[1][1]);
  }

  const int hi = lane >> 4, ln = lane & 15;
  #pragma unroll
  for (int mi = 0; mi < 2; ++mi) {
    #pragma unroll
    for (int nj = 0; nj < 2; ++nj) {
      int n = n0 + wn + nj * 16 + ln;
      #pragma unroll
      for (int j = 0; j < 8; ++j) {
        int m = m0 + wm + mi * 16 + j + hi * 8;
        out[(size_t)m * D_MODEL + n] = acc[mi][nj][j];
      }
    }
  }
}

// ---------------------------------------------------------------------------
extern "C" void kernel_launch(void* const* d_in, const int* in_sizes, int n_in,
                              void* d_out, int out_size, void* d_ws, size_t ws_size,
                              hipStream_t stream) {
  const float* kv   = (const float*)d_in[0];
  const float* q    = (const float*)d_in[1];
  const float* mask = (const float*)d_in[2];
  const float* WQ   = (const float*)d_in[3];
  const float* WK   = (const float*)d_in[4];
  const float* WV   = (const float*)d_in[5];
  const float* WO   = (const float*)d_in[6];
  const float* rb   = (const float*)d_in[7];
  float* out = (float*)d_out;

  char* ws = (char*)d_ws;
  size_t off = 0;
  auto alloc = [&](size_t bytes) -> void* {
    void* p = ws + off;
    off = (off + bytes + 255) & ~(size_t)255;
    return p;
  };
  __bf16* WQt  = (__bf16*)alloc((size_t)INNER * D_MODEL * 2);
  __bf16* WKt  = (__bf16*)alloc((size_t)INNER * D_MODEL * 2);
  __bf16* WVt  = (__bf16*)alloc((size_t)INNER * D_MODEL * 2);
  __bf16* WOt  = (__bf16*)alloc((size_t)D_MODEL * INNER * 2);
  float*  bias = (float*)alloc((size_t)N_HEADS * BIAS_LEN * 4);
  __bf16* Qb   = (__bf16*)alloc((size_t)BATCH * N_HEADS * SEQ * D_KV * 2);
  __bf16* Kb   = (__bf16*)alloc((size_t)BATCH * N_HEADS * SEQ * D_KV * 2);
  __bf16* Vb   = (__bf16*)alloc((size_t)BATCH * N_HEADS * SEQ * D_KV * 2);
  __bf16* AO   = (__bf16*)alloc((size_t)BATCH * SEQ * INNER * 2);

  k_transpose_bf16<<<512, 256, 0, stream>>>(WQ, WQt, D_MODEL, INNER);
  k_transpose_bf16<<<512, 256, 0, stream>>>(WK, WKt, D_MODEL, INNER);
  k_transpose_bf16<<<512, 256, 0, stream>>>(WV, WVt, D_MODEL, INNER);
  k_transpose_bf16<<<512, 256, 0, stream>>>(WO, WOt, INNER, D_MODEL);
  k_build_bias<<<(N_HEADS * BIAS_LEN + 255) / 256, 256, 0, stream>>>(rb, bias);

  k_proj_qkv<<<dim3((BATCH * SEQ) / 128, INNER / 64, 3), 256, 0, stream>>>(
      q, kv, WQt, WKt, WVt, Qb, Kb, Vb);

  k_attn<<<dim3(SEQ / 64, N_HEADS, BATCH), 128, 0, stream>>>(
      Qb, Kb, Vb, mask, bias, AO);

  k_proj_out<<<dim3((BATCH * SEQ) / 128, D_MODEL / 64), 256, 0, stream>>>(AO, WOt, out);
}
